// V2E_layer_set_27393301414291
// MI455X (gfx1250) — matrix operand order repaired and checked by reading, not compiled, not run on hardware
//
#include <hip/hip_runtime.h>
#include <hip/hip_bf16.h>
#include <math.h>

// ---------------------------------------------------------------------------
// V2E hypergraph layer for MI455X (gfx1250, wave32, WMMA f32_16x16x32_f16)
// Pipeline:
//   pack weights -> WMMA GEMM x=relu(node@Wv2e+b) [B in LDS, M=32/wave]
//   -> WMMA GEMM [xV|alpha] [B in LDS, M=32/wave]
//   -> segment softmax (atomic max / atomic add) -> weighted segment scatter
//   -> LN0 -> WMMA GEMM rff(+relu) -> LN1(+relu) -> dual WMMA GEMM (cat @ W_upd)
// ---------------------------------------------------------------------------

typedef __attribute__((ext_vector_type(16))) _Float16 v16h;
typedef __attribute__((ext_vector_type(8)))  _Float16 v8h;
typedef __attribute__((ext_vector_type(8)))  float    v8f;
typedef __attribute__((ext_vector_type(4)))  float    v4f;

#define DM 256
#define NHEADS 8
#define NEG_SLOPE 0.2f
#define LN_EPS 1e-5f

// ---- WMMA fragment loaders -------------------------------------------------
// A-matrix 16x32 f16 layout (wave32): m = lane&15, kh = lane>>4
//   halves 0..7  -> K = t*32 + kh*8 + i
//   halves 8..15 -> K = t*32 + 16 + kh*8 + i
static __device__ __forceinline__ v16h load_a_f16(const _Float16* rowp, int t, int kh) {
  const _Float16* p = rowp + t * 32 + kh * 8;
  v8h c1 = *(const v8h*)p;
  v8h c2 = *(const v8h*)(p + 16);
  v16h a;
#pragma unroll
  for (int i = 0; i < 8; ++i) { a[i] = c1[i]; a[i + 8] = c2[i]; }
  return a;
}

static __device__ __forceinline__ v16h load_a_f32(const float* rowp, int t, int kh) {
  const float* p = rowp + t * 32 + kh * 8;
  v4f c0 = *(const v4f*)p;
  v4f c1 = *(const v4f*)(p + 4);
  v4f c2 = *(const v4f*)(p + 16);
  v4f c3 = *(const v4f*)(p + 20);
  v16h a;
#pragma unroll
  for (int i = 0; i < 4; ++i) {
    a[i]      = (_Float16)c0[i];
    a[4 + i]  = (_Float16)c1[i];
    a[8 + i]  = (_Float16)c2[i];
    a[12 + i] = (_Float16)c3[i];
  }
  return a;
}

// B fragments pre-packed lane-major: fragment (nt,t) = 32 lanes x 16 halves.
static __device__ __forceinline__ v16h load_b(const _Float16* __restrict__ Bp,
                                              int nt, int t, int lane) {
  return *(const v16h*)(Bp + ((((size_t)nt * 8 + t) * 32 + lane) << 4));
}

// Same layout, but from LDS (ds_load_b128 x2 per fragment).
static __device__ __forceinline__ v16h load_b_lds(const _Float16* sh,
                                                  int nt, int t, int lane) {
  return *(const v16h*)(sh + ((((unsigned)nt * 8 + t) * 32 + lane) << 4));
}

// Cooperative LDS fill: copy `halves` _Float16 elements as 16B chunks.
static __device__ __forceinline__ void fill_lds(
    _Float16* sh, const _Float16* __restrict__ src, int halves, int tid, int nthr) {
  v4f* d = (v4f*)sh;
  const v4f* s = (const v4f*)src;
  const int chunks = halves >> 3;   // 8 halves = 16 bytes
  for (int i = tid; i < chunks; i += nthr) d[i] = s[i];
  __syncthreads();
}

// ---- weight packing --------------------------------------------------------
// Packs W[k][n] (k-major, ld=DM) into WMMA B-fragment order:
//   frag = nt*8 + t ; lane -> n = nt*16 + (lane&15) ; halves e -> K = t*32 + (lane>>4)*8 + e + (e>=8 ? 8 : 0)
__global__ void pack_b_kernel(const float* __restrict__ W, int ntiles,
                              _Float16* __restrict__ out) {
  int id = blockIdx.x * blockDim.x + threadIdx.x;
  if (id >= ntiles * 8 * 32) return;
  int lane = id & 31, frag = id >> 5;
  int t = frag & 7, nt = frag >> 3;
  int n = (nt << 4) + (lane & 15);
  int kbase = t * 32 + ((lane >> 4) << 3);
  _Float16* o = out + ((size_t)frag * 32 + lane) * 16;
#pragma unroll
  for (int e = 0; e < 16; ++e) o[e] = (_Float16)W[(size_t)(kbase + e + ((e >> 3) << 3)) * DM + n];
}

// Fills N-tile 16 of Wp_Vx with Wk_att[k][h] = sum_c W_K[k][h*32+c]*att_r[h,c]
// (zeros for h>=8), and biasVx[272] = [b_V ; b_alpha ; 0].
__global__ void pack_katt_kernel(const float* __restrict__ WK, const float* __restrict__ bK,
                                 const float* __restrict__ bV, const float* __restrict__ attr,
                                 _Float16* __restrict__ WpVx, float* __restrict__ biasVx) {
  int id = blockIdx.x * blockDim.x + threadIdx.x;
  if (id < 256) {
    int lane = id & 31, t = id >> 5;
    int nl = lane & 15;
    int kbase = t * 32 + ((lane >> 4) << 3);
    _Float16* o = WpVx + ((size_t)(16 * 8 + t) * 32 + lane) * 16;
    for (int e = 0; e < 16; ++e) {
      float v = 0.f;
      if (nl < NHEADS) {
        const float* wr = WK + (size_t)(kbase + e + ((e >> 3) << 3)) * DM + nl * 32;
        const float* ar = attr + nl * 32;
#pragma unroll
        for (int c = 0; c < 32; ++c) v += wr[c] * ar[c];
      }
      o[e] = (_Float16)v;
    }
  } else {
    int j = id - 256;
    if (j < 272) {
      float v = 0.f;
      if (j < 256) v = bV[j];
      else if (j < 264) {
        int h = j - 256;
#pragma unroll
        for (int c = 0; c < 32; ++c) v += bK[h * 32 + c] * attr[h * 32 + c];
      }
      biasVx[j] = v;
    }
  }
}

__global__ void cvt16_kernel(const float* __restrict__ in, _Float16* __restrict__ out,
                             size_t total) {
  size_t idx = (size_t)blockIdx.x * blockDim.x + threadIdx.x;
  if (idx < total) out[idx] = (_Float16)in[idx];
}

__global__ void init_kernel(float* __restrict__ outacc, unsigned* __restrict__ amax,
                            float* __restrict__ denom, int E) {
  int idx = blockIdx.x * 256 + threadIdx.x;  // grid = E*256 threads exactly
  outacc[idx] = 0.f;
  if (idx < E * NHEADS) { amax[idx] = 0u; denom[idx] = 0.f; }
}

// ---- big GEMMs: B in LDS, 32 rows per wave (each B fragment feeds 2 WMMAs) -
__global__ __launch_bounds__(128) void gemm_x_kernel(
    const float* __restrict__ A, const _Float16* __restrict__ Bp,
    const float* __restrict__ bias, _Float16* __restrict__ X, int rtiles32) {
  __shared__ _Float16 Bl[16 * 8 * 32 * 16];        // 128 KB packed W_v2e
  fill_lds(Bl, Bp, 16 * 8 * 32 * 16, threadIdx.x, 128);
  const int lane = threadIdx.x & 31, wave = threadIdx.x >> 5;
  const int m = lane & 15, kh = lane >> 4;
  for (int tile = blockIdx.x * 4 + wave; tile < rtiles32; tile += gridDim.x * 4) {
    const int row0 = tile << 5;
    const float* rp0 = A + (size_t)(row0 + m) * DM;
    const float* rp1 = A + (size_t)(row0 + 16 + m) * DM;
    v16h a0[8], a1[8];
#pragma unroll
    for (int t = 0; t < 8; ++t) { a0[t] = load_a_f32(rp0, t, kh); a1[t] = load_a_f32(rp1, t, kh); }
    const int tnext = tile + gridDim.x * 4;       // overlap next A tile with WMMAs
    if (tnext < rtiles32) {
      __builtin_prefetch(A + (size_t)((tnext << 5) + m) * DM, 0, 1);
      __builtin_prefetch(A + (size_t)((tnext << 5) + 16 + m) * DM, 0, 1);
    }
#pragma unroll 2
    for (int nt = 0; nt < 16; ++nt) {
      v8f c0 = {}, c1 = {};
#pragma unroll
      for (int t = 0; t < 8; ++t) {
        const v16h b = load_b_lds(Bl, nt, t, lane);
        c0 = __builtin_amdgcn_wmma_f32_16x16x32_f16(false, a0[t], false, b, (short)0, c0, false, false);
        c1 = __builtin_amdgcn_wmma_f32_16x16x32_f16(false, a1[t], false, b, (short)0, c1, false, false);
      }
      const int col = (nt << 4) + m;
      const float bv = bias[col];
      _Float16* d0 = X + (size_t)(row0 + (kh << 3)) * DM + col;
      _Float16* d1 = X + (size_t)(row0 + 16 + (kh << 3)) * DM + col;
#pragma unroll
      for (int r = 0; r < 8; ++r) {
        float v0 = c0[r] + bv, v1 = c1[r] + bv;
        d0[(size_t)r * DM] = (_Float16)(v0 > 0.f ? v0 : 0.f);   // relu
        d1[(size_t)r * DM] = (_Float16)(v1 > 0.f ? v1 : 0.f);
      }
    }
  }
}

// x -> [ xV (16 N-tiles) | alpha (tile 16, cols 0..7, leaky-relu) ]
__global__ __launch_bounds__(128) void gemm_v_kernel(
    const _Float16* __restrict__ Xh, const _Float16* __restrict__ Bp,
    const float* __restrict__ biasVx, _Float16* __restrict__ XV,
    float* __restrict__ alpha, int rtiles32) {
  __shared__ _Float16 Bl[17 * 8 * 32 * 16];        // 136 KB packed [W_V | Wk_att]
  fill_lds(Bl, Bp, 17 * 8 * 32 * 16, threadIdx.x, 128);
  const int lane = threadIdx.x & 31, wave = threadIdx.x >> 5;
  const int m = lane & 15, kh = lane >> 4;
  for (int tile = blockIdx.x * 4 + wave; tile < rtiles32; tile += gridDim.x * 4) {
    const int row0 = tile << 5;
    const _Float16* rp0 = Xh + (size_t)(row0 + m) * DM;
    const _Float16* rp1 = Xh + (size_t)(row0 + 16 + m) * DM;
    v16h a0[8], a1[8];
#pragma unroll
    for (int t = 0; t < 8; ++t) { a0[t] = load_a_f16(rp0, t, kh); a1[t] = load_a_f16(rp1, t, kh); }
    const int tnext = tile + gridDim.x * 4;
    if (tnext < rtiles32) {
      __builtin_prefetch(Xh + (size_t)((tnext << 5) + m) * DM, 0, 1);
      __builtin_prefetch(Xh + (size_t)((tnext << 5) + 16 + m) * DM, 0, 1);
    }
#pragma unroll 2
    for (int nt = 0; nt < 17; ++nt) {
      v8f c0 = {}, c1 = {};
#pragma unroll
      for (int t = 0; t < 8; ++t) {
        const v16h b = load_b_lds(Bl, nt, t, lane);
        c0 = __builtin_amdgcn_wmma_f32_16x16x32_f16(false, a0[t], false, b, (short)0, c0, false, false);
        c1 = __builtin_amdgcn_wmma_f32_16x16x32_f16(false, a1[t], false, b, (short)0, c1, false, false);
      }
      if (nt < 16) {
        const int col = (nt << 4) + m;
        const float bv = biasVx[col];
        _Float16* d0 = XV + (size_t)(row0 + (kh << 3)) * DM + col;
        _Float16* d1 = XV + (size_t)(row0 + 16 + (kh << 3)) * DM + col;
#pragma unroll
        for (int r = 0; r < 8; ++r) {
          d0[(size_t)r * DM] = (_Float16)(c0[r] + bv);
          d1[(size_t)r * DM] = (_Float16)(c1[r] + bv);
        }
      } else if (m < NHEADS) {
        const float bv = biasVx[256 + m];
        float* p0 = alpha + (size_t)(row0 + (kh << 3)) * NHEADS + m;
        float* p1 = alpha + (size_t)(row0 + 16 + (kh << 3)) * NHEADS + m;
#pragma unroll
        for (int r = 0; r < 8; ++r) {
          float v0 = c0[r] + bv, v1 = c1[r] + bv;
          p0[(size_t)r * NHEADS] = (v0 > 0.f) ? v0 : NEG_SLOPE * v0;   // leaky relu
          p1[(size_t)r * NHEADS] = (v1 > 0.f) ? v1 : NEG_SLOPE * v1;
        }
      }
    }
  }
}

// ---- small edge-side GEMMs keep streaming B from L2 (weights stay hot) -----
__global__ __launch_bounds__(128) void gemm_rff_kernel(
    const _Float16* __restrict__ Ah, const _Float16* __restrict__ Bp,
    const float* __restrict__ bias, float* __restrict__ Out, int rtiles) {
  const int lane = threadIdx.x & 31, wave = threadIdx.x >> 5;
  const int m = lane & 15, kh = lane >> 4;
  for (int tile = blockIdx.x * 4 + wave; tile < rtiles; tile += gridDim.x * 4) {
    const int row0 = tile << 4;
    const _Float16* rowp = Ah + (size_t)(row0 + m) * DM;
    v16h afr[8];
#pragma unroll
    for (int t = 0; t < 8; ++t) afr[t] = load_a_f16(rowp, t, kh);
#pragma unroll 4
    for (int nt = 0; nt < 16; ++nt) {
      v8f acc = {};
#pragma unroll
      for (int t = 0; t < 8; ++t)
        acc = __builtin_amdgcn_wmma_f32_16x16x32_f16(
            false, afr[t], false, load_b(Bp, nt, t, lane), (short)0, acc, false, false);
      const int col = (nt << 4) + m;
      const float bv = bias[col];
      float* dp = Out + (size_t)(row0 + (kh << 3)) * DM + col;
#pragma unroll
      for (int r = 0; r < 8; ++r) {
        float v = acc[r] + bv;
        dp[(size_t)r * DM] = v > 0.f ? v : 0.f;                 // relu
      }
    }
  }
}

// out = relu( t3 @ W_updT + hyperedge @ W_updB + b_upd )
__global__ __launch_bounds__(128) void gemm_upd_kernel(
    const _Float16* __restrict__ A1, const _Float16* __restrict__ Bp1,
    const _Float16* __restrict__ A2, const _Float16* __restrict__ Bp2,
    const float* __restrict__ bias, float* __restrict__ Out, int rtiles) {
  const int lane = threadIdx.x & 31, wave = threadIdx.x >> 5;
  const int m = lane & 15, kh = lane >> 4;
  for (int tile = blockIdx.x * 4 + wave; tile < rtiles; tile += gridDim.x * 4) {
    const int row0 = tile << 4;
    const _Float16* r1 = A1 + (size_t)(row0 + m) * DM;
    const _Float16* r2 = A2 + (size_t)(row0 + m) * DM;
    v16h a1[8], a2[8];
#pragma unroll
    for (int t = 0; t < 8; ++t) { a1[t] = load_a_f16(r1, t, kh); a2[t] = load_a_f16(r2, t, kh); }
#pragma unroll 2
    for (int nt = 0; nt < 16; ++nt) {
      v8f acc = {};
#pragma unroll
      for (int t = 0; t < 8; ++t)
        acc = __builtin_amdgcn_wmma_f32_16x16x32_f16(
            false, a1[t], false, load_b(Bp1, nt, t, lane), (short)0, acc, false, false);
#pragma unroll
      for (int t = 0; t < 8; ++t)
        acc = __builtin_amdgcn_wmma_f32_16x16x32_f16(
            false, a2[t], false, load_b(Bp2, nt, t, lane), (short)0, acc, false, false);
      const int col = (nt << 4) + m;
      const float bv = bias[col];
      float* dp = Out + (size_t)(row0 + (kh << 3)) * DM + col;
#pragma unroll
      for (int r = 0; r < 8; ++r) {
        float v = acc[r] + bv;
        dp[(size_t)r * DM] = v > 0.f ? v : 0.f;                 // relu
      }
    }
  }
}

// ---- segment softmax -------------------------------------------------------
static __device__ __forceinline__ unsigned fenc(float f) {
  unsigned u = __float_as_uint(f);
  return (u & 0x80000000u) ? ~u : (u | 0x80000000u);
}
static __device__ __forceinline__ float fdec(unsigned e) {
  unsigned u = (e & 0x80000000u) ? (e & 0x7fffffffu) : ~e;
  return __uint_as_float(u);
}

__global__ void seg_max_kernel(const float* __restrict__ alpha, const int* __restrict__ seg,
                               unsigned* __restrict__ amax, int N) {
  int idx = blockIdx.x * blockDim.x + threadIdx.x;
  if (idx >= N * NHEADS) return;
  int n = idx >> 3, h = idx & 7;
  atomicMax(&amax[(size_t)seg[n] * NHEADS + h], fenc(alpha[idx]));
}

__global__ void seg_exp_kernel(const float* __restrict__ alpha, const int* __restrict__ seg,
                               const unsigned* __restrict__ amax, float* __restrict__ ex,
                               float* __restrict__ denom, int N) {
  int idx = blockIdx.x * blockDim.x + threadIdx.x;
  if (idx >= N * NHEADS) return;
  int n = idx >> 3, h = idx & 7;
  int e = seg[n];
  float ev = expf(alpha[idx] - fdec(amax[(size_t)e * NHEADS + h]));
  ex[idx] = ev;
  atomicAdd(&denom[(size_t)e * NHEADS + h], ev);
}

__global__ void scatter_v_kernel(const _Float16* __restrict__ xv, const float* __restrict__ ex,
                                 const float* __restrict__ denom, const int* __restrict__ seg,
                                 float* __restrict__ outacc) {
  size_t idx = (size_t)blockIdx.x * 256 + threadIdx.x;   // grid = N blocks x 256
  int n = (int)(idx >> 8), c = (int)(idx & 255), h = c >> 5;
  int e = seg[n];
  float w = ex[(size_t)n * NHEADS + h] / fmaxf(denom[(size_t)e * NHEADS + h], 1e-16f);
  atomicAdd(&outacc[(size_t)e * DM + c], (float)xv[idx] * w);
}

// ---- layernorms (one row per 256-thread block) -----------------------------
__global__ __launch_bounds__(256) void ln0_kernel(
    const float* __restrict__ acc, const float* __restrict__ attr,
    const float* __restrict__ g, const float* __restrict__ b,
    float* __restrict__ t0f, _Float16* __restrict__ t0h) {
  __shared__ float s1[256], s2[256];
  const int i = threadIdx.x;
  const size_t base = (size_t)blockIdx.x * DM;
  float v = acc[base + i] + attr[i];                  // + seed residual att_r
  s1[i] = v; s2[i] = v * v;
  __syncthreads();
  for (int off = 128; off > 0; off >>= 1) {
    if (i < off) { s1[i] += s1[i + off]; s2[i] += s2[i + off]; }
    __syncthreads();
  }
  float mean = s1[0] * (1.f / DM);
  float var = s2[0] * (1.f / DM) - mean * mean;
  float rs = rsqrtf(var + LN_EPS);
  float o = (v - mean) * rs * g[i] + b[i];
  t0f[base + i] = o;
  t0h[base + i] = (_Float16)o;
}

__global__ __launch_bounds__(256) void ln1_kernel(
    const float* __restrict__ t0f, const float* __restrict__ rb,
    const float* __restrict__ g, const float* __restrict__ b,
    _Float16* __restrict__ t3h) {
  __shared__ float s1[256], s2[256];
  const int i = threadIdx.x;
  const size_t base = (size_t)blockIdx.x * DM;
  float v = t0f[base + i] + rb[base + i];             // rb already relu'd
  s1[i] = v; s2[i] = v * v;
  __syncthreads();
  for (int off = 128; off > 0; off >>= 1) {
    if (i < off) { s1[i] += s1[i + off]; s2[i] += s2[i + off]; }
    __syncthreads();
  }
  float mean = s1[0] * (1.f / DM);
  float var = s2[0] * (1.f / DM) - mean * mean;
  float rs = rsqrtf(var + LN_EPS);
  float o = (v - mean) * rs * g[i] + b[i];
  t3h[base + i] = (_Float16)(o > 0.f ? o : 0.f);      // relu -> hyperedge_tem
}

// ---------------------------------------------------------------------------
extern "C" void kernel_launch(void* const* d_in, const int* in_sizes, int n_in,
                              void* d_out, int out_size, void* d_ws, size_t ws_size,
                              hipStream_t stream) {
  const float* hyperedge  = (const float*)d_in[0];
  const float* hyper_node = (const float*)d_in[1];
  const int*   ve         = (const int*)d_in[2];
  const float* W_v2e = (const float*)d_in[3];
  const float* b_v2e = (const float*)d_in[4];
  const float* W_K   = (const float*)d_in[5];
  const float* b_K   = (const float*)d_in[6];
  const float* W_V   = (const float*)d_in[7];
  const float* b_V   = (const float*)d_in[8];
  const float* att_r = (const float*)d_in[9];
  const float* ln0_g = (const float*)d_in[10];
  const float* ln0_b = (const float*)d_in[11];
  const float* ln1_g = (const float*)d_in[12];
  const float* ln1_b = (const float*)d_in[13];
  const float* W_rff = (const float*)d_in[14];
  const float* b_rff = (const float*)d_in[15];
  const float* W_upd = (const float*)d_in[16];
  const float* b_upd = (const float*)d_in[17];

  const int N = in_sizes[1] / DM;   // 262144
  const int E = in_sizes[0] / DM;   // 16384
  const int* seg = ve;              // ve_affiliation[0]

  // workspace carve (all 256B aligned)
  char* p = (char*)d_ws;
  auto carve = [&](size_t bytes) -> char* {
    char* r = p; p += (bytes + 255) & ~(size_t)255; return r;
  };
  const size_t packB = (size_t)16 * 8 * 32 * 16 * sizeof(_Float16);   // 128KB
  _Float16* Wp_v2e = (_Float16*)carve(packB);
  _Float16* Wp_Vx  = (_Float16*)carve((size_t)17 * 8 * 32 * 16 * sizeof(_Float16));
  _Float16* Wp_rff = (_Float16*)carve(packB);
  _Float16* Wp_uT  = (_Float16*)carve(packB);
  _Float16* Wp_uB  = (_Float16*)carve(packB);
  float*    biasVx = (float*)carve(272 * sizeof(float));
  _Float16* he16   = (_Float16*)carve((size_t)E * DM * sizeof(_Float16));
  _Float16* x16    = (_Float16*)carve((size_t)N * DM * sizeof(_Float16));
  _Float16* xv16   = (_Float16*)carve((size_t)N * DM * sizeof(_Float16));
  float*    alpha  = (float*)carve((size_t)N * NHEADS * sizeof(float));
  float*    ex     = (float*)carve((size_t)N * NHEADS * sizeof(float));
  unsigned* amax   = (unsigned*)carve((size_t)E * NHEADS * sizeof(unsigned));
  float*    denom  = (float*)carve((size_t)E * NHEADS * sizeof(float));
  float*    outacc = (float*)carve((size_t)E * DM * sizeof(float));
  float*    t0f    = (float*)carve((size_t)E * DM * sizeof(float));
  _Float16* t0h    = (_Float16*)carve((size_t)E * DM * sizeof(_Float16));
  float*    rbuf   = (float*)carve((size_t)E * DM * sizeof(float));
  _Float16* t3h    = (_Float16*)carve((size_t)E * DM * sizeof(_Float16));
  (void)ws_size; (void)n_in; (void)out_size;

  // 1) pack weights into WMMA B-fragment layout (tiny; L2 resident afterwards)
  pack_b_kernel<<<16, 256, 0, stream>>>(W_v2e, 16, Wp_v2e);
  pack_b_kernel<<<16, 256, 0, stream>>>(W_V, 16, Wp_Vx);
  pack_b_kernel<<<16, 256, 0, stream>>>(W_rff, 16, Wp_rff);
  pack_b_kernel<<<16, 256, 0, stream>>>(W_upd, 16, Wp_uT);
  pack_b_kernel<<<16, 256, 0, stream>>>(W_upd + (size_t)256 * DM, 16, Wp_uB);
  pack_katt_kernel<<<3, 256, 0, stream>>>(W_K, b_K, b_V, att_r, Wp_Vx, biasVx);
  cvt16_kernel<<<(int)(((size_t)E * DM + 255) / 256), 256, 0, stream>>>(hyperedge, he16,
                                                                        (size_t)E * DM);
  init_kernel<<<E, 256, 0, stream>>>(outacc, amax, denom, E);

  // 2) big WMMA GEMMs (B in LDS, 32 rows per wave)
  gemm_x_kernel<<<512, 128, 0, stream>>>(hyper_node, Wp_v2e, b_v2e, x16, N / 32);
  gemm_v_kernel<<<512, 128, 0, stream>>>(x16, Wp_Vx, biasVx, xv16, alpha, N / 32);

  // 3) segment softmax + weighted segment sum (random segments -> L2 atomics)
  int segBlocks = (N * NHEADS + 255) / 256;
  seg_max_kernel<<<segBlocks, 256, 0, stream>>>(alpha, seg, amax, N);
  seg_exp_kernel<<<segBlocks, 256, 0, stream>>>(alpha, seg, amax, ex, denom, N);
  scatter_v_kernel<<<N, 256, 0, stream>>>(xv16, ex, denom, seg, outacc);

  // 4) edge-side: LN0 -> rFF GEMM -> LN1 -> concat GEMM -> d_out
  ln0_kernel<<<E, 256, 0, stream>>>(outacc, att_r, ln0_g, ln0_b, t0f, t0h);
  gemm_rff_kernel<<<256, 128, 0, stream>>>(t0h, Wp_rff, b_rff, rbuf, E / 16);
  ln1_kernel<<<E, 256, 0, stream>>>(t0f, rbuf, ln1_g, ln1_b, t3h);
  gemm_upd_kernel<<<256, 128, 0, stream>>>(t3h, Wp_uT, he16, Wp_uB, b_upd,
                                           (float*)d_out, E / 16);
}
